// MRCGNN_27066883899440
// MI455X (gfx1250) — matrix operand order repaired and checked by reading, not compiled, not run on hardware
//
#include <hip/hip_runtime.h>
#include <math.h>

typedef __attribute__((ext_vector_type(16))) _Float16 v16h;
typedef __attribute__((ext_vector_type(8)))  float    v8f;

#define FIN 128
#define H1D 64
#define H2D 32

// ---------------------------------------------------------------- init: zero floats, bins = -1, ints = 0
__global__ void init_kernel(float* fz, long nf, int* bins, long nb,
                            int* iz, int ni, float* csum) {
  long stride = (long)gridDim.x * blockDim.x;
  long i0 = (long)blockIdx.x * blockDim.x + threadIdx.x;
  for (long i = i0; i < nf; i += stride) fz[i] = 0.0f;
  for (long i = i0; i < nb; i += stride) bins[i] = -1;
  if (blockIdx.x == 0) {
    for (int i = threadIdx.x; i < ni; i += blockDim.x) iz[i] = 0;
    if (threadIdx.x < 32) csum[threadIdx.x] = 0.0f;
  }
}

// ---------------------------------------------------------------- edge counts + relation histograms
__global__ void count_kernel(const int* __restrict__ dst,
                             const int* __restrict__ etA,
                             const int* __restrict__ etB,
                             float* cntA, float* cntB,
                             int* histA, int* histB, int E, int N) {
  int e = blockIdx.x * blockDim.x + threadIdx.x;
  if (e >= E) return;
  int rA = etA[e], rB = etB[e], d = dst[e];
  atomicAdd(&cntA[(long)rA * N + d], 1.0f);
  atomicAdd(&cntB[(long)rB * N + d], 1.0f);
  atomicAdd(&histA[rA], 1);
  atomicAdd(&histB[rB], 1);
}

// ---------------------------------------------------------------- padded exclusive scan over 65 relations
__global__ void scan_kernel(const int* histA, const int* histB,
                            int* baseA, int* baseB, int R) {
  if (threadIdx.x == 0) {
    int a = 0, b = 0;
    for (int r = 0; r < R; ++r) {
      baseA[r] = a; a += (histA[r] + 15) & ~15;
      baseB[r] = b; b += (histB[r] + 15) & ~15;
    }
    baseA[R] = a; baseB[R] = b;
  }
}

// ---------------------------------------------------------------- scatter edge ids into relation bins
__global__ void fill_kernel(const int* __restrict__ etA, const int* __restrict__ etB,
                            int* curA, int* curB,
                            const int* __restrict__ baseA, const int* __restrict__ baseB,
                            int* binA, int* binB, int E) {
  int e = blockIdx.x * blockDim.x + threadIdx.x;
  if (e >= E) return;
  int rA = etA[e];
  int p = atomicAdd(&curA[rA], 1);
  binA[baseA[rA] + p] = e;
  int rB = etB[e];
  int q = atomicAdd(&curB[rB], 1);
  binB[baseB[rB] + q] = e;
}

// ---------------------------------------------------------------- WMMA dense GEMM
// C[M,Ncols] = A[M,K] * B[K,Ncols] (+bias). One wave per 16x16 tile. M%16==0, K%32==0.
__global__ __launch_bounds__(32)
void wmma_gemm_kernel(const float* __restrict__ A, const float* __restrict__ Bw,
                      float* __restrict__ C, const float* __restrict__ bias,
                      int M, int K, int Ncols, int ldc) {
  __shared__ float As[16 * 32];
  __shared__ float Bs[32 * 16];
  const int lane   = threadIdx.x;
  const int mt     = blockIdx.x;
  const int nt     = blockIdx.y;
  const int mrow   = lane & 15;
  const int khalfA = (lane >> 4) * 8;
  const int kbaseB = (lane >> 4) * 16;
  const int ncol   = lane & 15;

  v8f acc = {};
  for (int ks = 0; ks < K; ks += 32) {
#pragma unroll
    for (int i = 0; i < 16; ++i) {
      int idx = i * 32 + lane;
      int ra = idx >> 5, ca = idx & 31;
      As[idx] = A[(long)(mt * 16 + ra) * K + ks + ca];
      int rb = idx >> 4, cb = idx & 15;
      int gc = nt * 16 + cb;
      Bs[idx] = (gc < Ncols) ? Bw[(long)(ks + rb) * Ncols + gc] : 0.0f;
    }
    __syncthreads();
    v16h a, b;
#pragma unroll
    for (int v = 0; v < 8; ++v) {
      int base = 2 * khalfA + 2 * v;
      a[2 * v]     = (_Float16)As[mrow * 32 + base];
      a[2 * v + 1] = (_Float16)As[mrow * 32 + base + 1];
      b[2 * v]     = (_Float16)Bs[(kbaseB + 2 * v) * 16 + ncol];
      b[2 * v + 1] = (_Float16)Bs[(kbaseB + 2 * v + 1) * 16 + ncol];
    }
    acc = __builtin_amdgcn_wmma_f32_16x16x32_f16(false, a, false, b,
                                                 (short)0, acc, false, false);
    __syncthreads();
  }
  int gcol = nt * 16 + ncol;
  if (gcol < Ncols) {
    float bv = bias ? bias[gcol] : 0.0f;
    int rbase = (lane >> 4) * 8;
#pragma unroll
    for (int v = 0; v < 8; ++v)
      C[(long)(mt * 16 + rbase + v) * ldc + gcol] = acc[v] + bv;
  }
}

// ---------------------------------------------------------------- binned WMMA edge transform, layer 1
// One wave per 16-edge tile of a single relation: msg(16x64) = Ahat(16x128) * W1[r],
// where Ahat rows are x[src]/cnt(r,dst); rows atomically scattered to acc[dst].
template <int DUAL>
__global__ __launch_bounds__(32)
void edge_l1_wmma_kernel(const int* __restrict__ bin, const int* __restrict__ base,
                         const int* __restrict__ src, const int* __restrict__ dst,
                         const float* __restrict__ xA, const float* __restrict__ xB,
                         const float* __restrict__ W1, const float* __restrict__ cnt,
                         float* accA, float* accB, int N, int R) {
  __shared__ _Float16 lxA[16 * FIN];
  __shared__ _Float16 lxB[DUAL ? 16 * FIN : 16];
  __shared__ int   lsrc[16];
  __shared__ int   ldst[16];
  __shared__ float linv[16];
  const int lane = threadIdx.x;
  const int t = blockIdx.x;
  if (t >= (base[R] >> 4)) return;
  const int t16 = t << 4;
  int r = 0;
  while (base[r + 1] <= t16) ++r;       // 65-entry scalar scan

  if (lane < 16) {
    int eid = bin[t16 + lane];
    int s = -1, d = 0;
    float inv = 0.0f;
    if (eid >= 0) {
      s = src[eid];
      d = dst[eid];
      inv = 1.0f / fmaxf(cnt[(long)r * N + d], 1.0f);
    }
    lsrc[lane] = s; ldst[lane] = d; linv[lane] = inv;
  }
  __syncthreads();

  // stage scaled f16 A tiles (lane covers 4 consecutive columns per row)
  const int c0 = lane * 4;
  for (int row = 0; row < 16; ++row) {
    int s = lsrc[row];
    float inv = linv[row];
    float4 va = {0.f, 0.f, 0.f, 0.f}, vb = {0.f, 0.f, 0.f, 0.f};
    if (s >= 0) {
      va = *(const float4*)(xA + (long)s * FIN + c0);
      if (DUAL) vb = *(const float4*)(xB + (long)s * FIN + c0);
    }
    lxA[row * FIN + c0 + 0] = (_Float16)(va.x * inv);
    lxA[row * FIN + c0 + 1] = (_Float16)(va.y * inv);
    lxA[row * FIN + c0 + 2] = (_Float16)(va.z * inv);
    lxA[row * FIN + c0 + 3] = (_Float16)(va.w * inv);
    if (DUAL) {
      lxB[row * FIN + c0 + 0] = (_Float16)(vb.x * inv);
      lxB[row * FIN + c0 + 1] = (_Float16)(vb.y * inv);
      lxB[row * FIN + c0 + 2] = (_Float16)(vb.z * inv);
      lxB[row * FIN + c0 + 3] = (_Float16)(vb.w * inv);
    }
  }
  __syncthreads();

  const int mrow  = lane & 15;
  const int khalf = (lane >> 4) * 8;
  const int kbase = (lane >> 4) * 16;
  const int ncol  = lane & 15;
  const float* Wr = W1 + (long)r * FIN * H1D;

  v8f aco[4] = {};
  v8f aca[4] = {};
  for (int ks = 0; ks < FIN; ks += 32) {
    v16h a0, a1;
#pragma unroll
    for (int v = 0; v < 8; ++v) {
      int bb = 2 * khalf + 2 * v;
      a0[2 * v]     = lxA[mrow * FIN + ks + bb];
      a0[2 * v + 1] = lxA[mrow * FIN + ks + bb + 1];
      if (DUAL) {
        a1[2 * v]     = lxB[mrow * FIN + ks + bb];
        a1[2 * v + 1] = lxB[mrow * FIN + ks + bb + 1];
      }
    }
#pragma unroll
    for (int nt = 0; nt < 4; ++nt) {
      v16h b;
#pragma unroll
      for (int i = 0; i < 16; ++i)
        b[i] = (_Float16)Wr[(ks + kbase + i) * H1D + nt * 16 + ncol];
      aco[nt] = __builtin_amdgcn_wmma_f32_16x16x32_f16(false, a0, false, b,
                                                       (short)0, aco[nt], false, false);
      if (DUAL)
        aca[nt] = __builtin_amdgcn_wmma_f32_16x16x32_f16(false, a1, false, b,
                                                         (short)0, aca[nt], false, false);
    }
  }

  int rb = (lane >> 4) * 8;
#pragma unroll
  for (int v = 0; v < 8; ++v) {
    int d = ldst[rb + v];
#pragma unroll
    for (int nt = 0; nt < 4; ++nt) {
      atomicAdd(&accA[(long)d * H1D + nt * 16 + ncol], aco[nt][v]);
      if (DUAL) atomicAdd(&accB[(long)d * H1D + nt * 16 + ncol], aca[nt][v]);
    }
  }
}

// ---------------------------------------------------------------- binned WMMA edge transform, layer 2
template <int DUAL>
__global__ __launch_bounds__(32)
void edge_l2_wmma_kernel(const int* __restrict__ bin, const int* __restrict__ base,
                         const int* __restrict__ src, const int* __restrict__ dst,
                         const float* __restrict__ hA, const float* __restrict__ hB,
                         const float* __restrict__ W2, const float* __restrict__ cnt,
                         float* accA, float* accB, int N, int R) {
  __shared__ _Float16 lhA[16 * H1D];
  __shared__ _Float16 lhB[DUAL ? 16 * H1D : 16];
  __shared__ int   lsrc[16];
  __shared__ int   ldst[16];
  __shared__ float linv[16];
  const int lane = threadIdx.x;
  const int t = blockIdx.x;
  if (t >= (base[R] >> 4)) return;
  const int t16 = t << 4;
  int r = 0;
  while (base[r + 1] <= t16) ++r;

  if (lane < 16) {
    int eid = bin[t16 + lane];
    int s = -1, d = 0;
    float inv = 0.0f;
    if (eid >= 0) {
      s = src[eid];
      d = dst[eid];
      inv = 1.0f / fmaxf(cnt[(long)r * N + d], 1.0f);
    }
    lsrc[lane] = s; ldst[lane] = d; linv[lane] = inv;
  }
  __syncthreads();

  const int c0 = lane * 2;              // 64 cols / 32 lanes
  for (int row = 0; row < 16; ++row) {
    int s = lsrc[row];
    float inv = linv[row];
    float2 va = {0.f, 0.f}, vb = {0.f, 0.f};
    if (s >= 0) {
      va = *(const float2*)(hA + (long)s * H1D + c0);
      if (DUAL) vb = *(const float2*)(hB + (long)s * H1D + c0);
    }
    lhA[row * H1D + c0 + 0] = (_Float16)(va.x * inv);
    lhA[row * H1D + c0 + 1] = (_Float16)(va.y * inv);
    if (DUAL) {
      lhB[row * H1D + c0 + 0] = (_Float16)(vb.x * inv);
      lhB[row * H1D + c0 + 1] = (_Float16)(vb.y * inv);
    }
  }
  __syncthreads();

  const int mrow  = lane & 15;
  const int khalf = (lane >> 4) * 8;
  const int kbase = (lane >> 4) * 16;
  const int ncol  = lane & 15;
  const float* Wr = W2 + (long)r * H1D * H2D;

  v8f aco[2] = {};
  v8f aca[2] = {};
  for (int ks = 0; ks < H1D; ks += 32) {
    v16h a0, a1;
#pragma unroll
    for (int v = 0; v < 8; ++v) {
      int bb = 2 * khalf + 2 * v;
      a0[2 * v]     = lhA[mrow * H1D + ks + bb];
      a0[2 * v + 1] = lhA[mrow * H1D + ks + bb + 1];
      if (DUAL) {
        a1[2 * v]     = lhB[mrow * H1D + ks + bb];
        a1[2 * v + 1] = lhB[mrow * H1D + ks + bb + 1];
      }
    }
#pragma unroll
    for (int nt = 0; nt < 2; ++nt) {
      v16h b;
#pragma unroll
      for (int i = 0; i < 16; ++i)
        b[i] = (_Float16)Wr[(ks + kbase + i) * H2D + nt * 16 + ncol];
      aco[nt] = __builtin_amdgcn_wmma_f32_16x16x32_f16(false, a0, false, b,
                                                       (short)0, aco[nt], false, false);
      if (DUAL)
        aca[nt] = __builtin_amdgcn_wmma_f32_16x16x32_f16(false, a1, false, b,
                                                         (short)0, aca[nt], false, false);
    }
  }

  int rb = (lane >> 4) * 8;
#pragma unroll
  for (int v = 0; v < 8; ++v) {
    int d = ldst[rb + v];
#pragma unroll
    for (int nt = 0; nt < 2; ++nt) {
      atomicAdd(&accA[(long)d * H2D + nt * 16 + ncol], aco[nt][v]);
      if (DUAL) atomicAdd(&accB[(long)d * H2D + nt * 16 + ncol], aca[nt][v]);
    }
  }
}

// ---------------------------------------------------------------- finalize: out = act(acc+rt+bias)
__global__ void add_bias_act_kernel(const float* __restrict__ acc,
                                    const float* __restrict__ rt,
                                    const float* __restrict__ bias,
                                    float* __restrict__ out,
                                    long n, int H, int relu) {
  long stride = (long)gridDim.x * blockDim.x;
  for (long i = (long)blockIdx.x * blockDim.x + threadIdx.x; i < n; i += stride) {
    float v = acc[i] + rt[i] + bias[(int)(i % H)];
    out[i] = relu ? fmaxf(v, 0.0f) : v;
  }
}

// ---------------------------------------------------------------- column sums of [N,32]
__global__ void colsum_kernel(const float* __restrict__ x, float* csum, long n) {
  long stride = (long)gridDim.x * blockDim.x;        // multiple of 32
  long i = (long)blockIdx.x * blockDim.x + threadIdx.x;
  int col = (int)(i & 31);
  float s = 0.0f;
  for (; i < n; i += stride) s += x[i];
  atomicAdd(&csum[col], s);
}

// ---------------------------------------------------------------- c = sigmoid(mean), w = disc_W @ c
__global__ void cw_kernel(const float* __restrict__ csum,
                          const float* __restrict__ discW,
                          float* wvec, int N) {
  __shared__ float c[32];
  int t = threadIdx.x;
  c[t] = 1.0f / (1.0f + expf(-csum[t] / (float)N));
  __syncthreads();
  float s = 0.0f;
#pragma unroll
  for (int j = 0; j < 32; ++j) s += discW[t * 32 + j] * c[j];
  wvec[t] = s;
}

// ---------------------------------------------------------------- discriminator scores
__global__ __launch_bounds__(256)
void disc_kernel(const float* __restrict__ x2o, const float* __restrict__ x2a,
                 const float* __restrict__ x2b, const float* __restrict__ wvec,
                 const float* __restrict__ disc_b,
                 float* ret_os, float* ret_os_a, int N) {
  int wave = threadIdx.x >> 5, lane = threadIdx.x & 31;
  int n = blockIdx.x * 8 + wave;
  if (n >= N) return;
  float wl = wvec[lane];
  float s1 = x2o[(long)n * 32 + lane] * wl;
  float s2 = x2a[(long)n * 32 + lane] * wl;
  float s3 = x2b[(long)n * 32 + lane] * wl;
#pragma unroll
  for (int off = 16; off > 0; off >>= 1) {
    s1 += __shfl_xor(s1, off, 32);
    s2 += __shfl_xor(s2, off, 32);
    s3 += __shfl_xor(s3, off, 32);
  }
  if (lane == 0) {
    float db = disc_b[0];
    ret_os[(long)n * 2 + 0]   = s1 + db;
    ret_os[(long)n * 2 + 1]   = s2 + db;
    ret_os_a[(long)n * 2 + 0] = s1 + db;
    ret_os_a[(long)n * 2 + 1] = s3 + db;
  }
}

// ---------------------------------------------------------------- build classifier input [B,448]
__global__ void gather_e_kernel(const float* __restrict__ x1o,
                                const float* __restrict__ x2o,
                                const float* __restrict__ feats,
                                const int* __restrict__ idx0,
                                const int* __restrict__ idx1,
                                const float* __restrict__ attt,
                                float* __restrict__ e, int B) {
  long i = (long)blockIdx.x * blockDim.x + threadIdx.x;
  long total = (long)B * 448;
  if (i >= total) return;
  int b = (int)(i / 448), col = (int)(i % 448);
  int half = (col >= 224);
  int c2 = half ? col - 224 : col;
  int node = half ? idx1[b] : idx0[b];
  float v;
  if (c2 < 64)       v = attt[0] * x1o[(long)node * 64 + c2];
  else if (c2 < 96)  v = attt[1] * x2o[(long)node * 32 + (c2 - 64)];
  else               v = feats[(long)node * 128 + (c2 - 96)];
  e[i] = v;
}

// ================================================================ launch
extern "C" void kernel_launch(void* const* d_in, const int* in_sizes, int n_in,
                              void* d_out, int out_size, void* d_ws, size_t ws_size,
                              hipStream_t stream) {
  const float* x_o       = (const float*)d_in[0];
  const float* x_a       = (const float*)d_in[1];
  const float* features1 = (const float*)d_in[2];
  const float* W1        = (const float*)d_in[3];
  const float* root1     = (const float*)d_in[4];
  const float* b1        = (const float*)d_in[5];
  const float* W2        = (const float*)d_in[6];
  const float* root2     = (const float*)d_in[7];
  const float* b2        = (const float*)d_in[8];
  const float* attt      = (const float*)d_in[9];
  const float* disc_W    = (const float*)d_in[10];
  const float* disc_b    = (const float*)d_in[11];
  const float* cls_W     = (const float*)d_in[12];
  const float* cls_b     = (const float*)d_in[13];
  const int*   edge_idx  = (const int*)d_in[14];
  const int*   e_type    = (const int*)d_in[15];
  const int*   e_type1   = (const int*)d_in[16];
  const int*   idxp      = (const int*)d_in[17];

  const int N = in_sizes[0] / FIN;   // 20000
  const int E = in_sizes[14] / 2;    // 640000
  const int B = in_sizes[17] / 2;    // 8192
  const int R = in_sizes[13];        // 65

  const int* src  = edge_idx;
  const int* dst  = edge_idx + E;
  const int* idx0 = idxp;
  const int* idx1 = idxp + B;

  const long maxPad   = ((long)E + 15L * R + 15L) & ~15L;   // padded bin capacity
  const int  maxTiles = (int)(maxPad >> 4);

  float* ws = (float*)d_ws;
  long off = 0;
  float* cntA   = ws + off; off += (long)R * N;
  float* cntB   = ws + off; off += (long)R * N;
  float* acc1_o = ws + off; off += (long)N * H1D;
  float* acc1_a = ws + off; off += (long)N * H1D;
  float* acc1_b = ws + off; off += (long)N * H1D;
  float* acc2_o = ws + off; off += (long)N * H2D;
  float* acc2_a = ws + off; off += (long)N * H2D;
  float* acc2_b = ws + off; off += (long)N * H2D;
  const long zero_nf = off;           // atomically-accumulated float region
  float* rt1_o  = ws + off; off += (long)N * H1D;
  float* rt1_a  = ws + off; off += (long)N * H1D;
  float* rt2_o  = ws + off; off += (long)N * H2D;
  float* rt2_a  = ws + off; off += (long)N * H2D;
  float* rt2_b  = ws + off; off += (long)N * H2D;
  float* emat   = ws + off; off += (long)B * 448;
  float* csum   = ws + off; off += 32;
  float* wvec   = ws + off; off += 32;
  int* histA = (int*)(ws + off); off += R;
  int* curA  = (int*)(ws + off); off += R;
  int* histB = (int*)(ws + off); off += R;
  int* curB  = (int*)(ws + off); off += R;
  const long int_zero_off = off - 4L * R;   // histA..curB contiguous, 4R ints
  int* baseA = (int*)(ws + off); off += R + 1;
  int* baseB = (int*)(ws + off); off += R + 1;
  int* binA  = (int*)(ws + off); off += maxPad;
  int* binB  = (int*)(ws + off); off += maxPad;
  (void)ws_size; (void)n_in; (void)out_size; (void)int_zero_off;

  float* out      = (float*)d_out;
  float* out_log  = out;                       // [B, R]
  float* out_ret  = out_log + (long)B * R;     // [N, 2]
  float* out_reta = out_ret + (long)N * 2;     // [N, 2]
  float* x2_o     = out_reta + (long)N * 2;    // [N, 32] final output slot, reused

  // 0) init: zero accumulators/counters, bins = -1
  init_kernel<<<2048, 256, 0, stream>>>(ws, zero_nf, binA, 2L * maxPad,
                                        histA, 4 * R, csum);

  // 1) per-(relation,dst) counts + per-relation histograms
  count_kernel<<<(E + 255) / 256, 256, 0, stream>>>(dst, e_type, e_type1,
                                                    cntA, cntB, histA, histB, E, N);

  // 2) padded prefix sums + bin fill
  scan_kernel<<<1, 32, 0, stream>>>(histA, histB, baseA, baseB, R);
  fill_kernel<<<(E + 255) / 256, 256, 0, stream>>>(e_type, e_type1, curA, curB,
                                                   baseA, baseB, binA, binB, E);

  // 3) root terms, layer 1 (WMMA)
  wmma_gemm_kernel<<<dim3(N / 16, H1D / 16), 32, 0, stream>>>(x_o, root1, rt1_o, nullptr, N, FIN, H1D, H1D);
  wmma_gemm_kernel<<<dim3(N / 16, H1D / 16), 32, 0, stream>>>(x_a, root1, rt1_a, nullptr, N, FIN, H1D, H1D);

  // 4) layer-1 binned WMMA edge aggregation (x_o/x_a fused on e_type; x_o on e_type1)
  edge_l1_wmma_kernel<1><<<maxTiles, 32, 0, stream>>>(binA, baseA, src, dst, x_o, x_a,
                                                      W1, cntA, acc1_o, acc1_a, N, R);
  edge_l1_wmma_kernel<0><<<maxTiles, 32, 0, stream>>>(binB, baseB, src, dst, x_o, nullptr,
                                                      W1, cntB, acc1_b, nullptr, N, R);

  // 5) h1 = relu(acc + root + b1)
  add_bias_act_kernel<<<2048, 256, 0, stream>>>(acc1_o, rt1_o, b1, acc1_o, (long)N * H1D, H1D, 1);
  add_bias_act_kernel<<<2048, 256, 0, stream>>>(acc1_a, rt1_a, b1, acc1_a, (long)N * H1D, H1D, 1);
  add_bias_act_kernel<<<2048, 256, 0, stream>>>(acc1_b, rt1_o, b1, acc1_b, (long)N * H1D, H1D, 1);

  // 6) root terms, layer 2 (WMMA)
  wmma_gemm_kernel<<<dim3(N / 16, H2D / 16), 32, 0, stream>>>(acc1_o, root2, rt2_o, nullptr, N, H1D, H2D, H2D);
  wmma_gemm_kernel<<<dim3(N / 16, H2D / 16), 32, 0, stream>>>(acc1_a, root2, rt2_a, nullptr, N, H1D, H2D, H2D);
  wmma_gemm_kernel<<<dim3(N / 16, H2D / 16), 32, 0, stream>>>(acc1_b, root2, rt2_b, nullptr, N, H1D, H2D, H2D);

  // 7) layer-2 binned WMMA edge aggregation
  edge_l2_wmma_kernel<1><<<maxTiles, 32, 0, stream>>>(binA, baseA, src, dst, acc1_o, acc1_a,
                                                      W2, cntA, acc2_o, acc2_a, N, R);
  edge_l2_wmma_kernel<0><<<maxTiles, 32, 0, stream>>>(binB, baseB, src, dst, acc1_b, nullptr,
                                                      W2, cntB, acc2_b, nullptr, N, R);

  // 8) x2 = acc + root + b2 (x2_o lands directly in d_out)
  add_bias_act_kernel<<<2048, 256, 0, stream>>>(acc2_o, rt2_o, b2, x2_o, (long)N * H2D, H2D, 0);
  add_bias_act_kernel<<<2048, 256, 0, stream>>>(acc2_a, rt2_a, b2, acc2_a, (long)N * H2D, H2D, 0);
  add_bias_act_kernel<<<2048, 256, 0, stream>>>(acc2_b, rt2_b, b2, acc2_b, (long)N * H2D, H2D, 0);

  // 9) readout c = sigmoid(mean(x2_o)), w = disc_W @ c
  colsum_kernel<<<80, 256, 0, stream>>>(x2_o, csum, (long)N * H2D);
  cw_kernel<<<1, 32, 0, stream>>>(csum, disc_W, wvec, N);

  // 10) discriminator scores
  disc_kernel<<<(N + 7) / 8, 256, 0, stream>>>(x2_o, acc2_a, acc2_b, wvec, disc_b,
                                               out_ret, out_reta, N);

  // 11) classifier input gather + logits GEMM (WMMA, bias fused)
  gather_e_kernel<<<(int)(((long)B * 448 + 255) / 256), 256, 0, stream>>>(
      acc1_o, x2_o, features1, idx0, idx1, attt, emat, B);
  wmma_gemm_kernel<<<dim3(B / 16, (R + 15) / 16), 32, 0, stream>>>(
      emat, cls_W, out_log, cls_b, B, 448, R, R);
}
